// GCN_25314537242763
// MI455X (gfx1250) — matrix operand order repaired and checked by reading, not compiled, not run on hardware
//
#include <hip/hip_runtime.h>
#include <math.h>

#define F_IN 1433
#define HID  67
#define NCLS 7
#define KPAD 1440   // 45 chunks of 32
#define NPAD 80     // 5 tiles of 16
#define ASTRIDE 36  // 32 f32 per row + 4 dword TDM pad
#define BSTRIDE 40  // 32 bf16 per row + 8 half (4 dword) TDM pad

typedef __attribute__((ext_vector_type(16))) __bf16 v16bf;
typedef __attribute__((ext_vector_type(8)))  float  v8f;
typedef __attribute__((ext_vector_type(8)))  unsigned short v8us;
typedef __attribute__((ext_vector_type(4)))  unsigned int v4u;
typedef __attribute__((ext_vector_type(8)))  int v8i;
typedef __attribute__((ext_vector_type(4)))  int v4i;

union BF16x16 { v16bf v; __bf16 h[16]; v8us q[2]; };

__device__ __forceinline__ __bf16 tobf(float f) { return (__bf16)f; }

__device__ __forceinline__ unsigned short f2bf_bits(float f) {
    union { float f; unsigned u; } x; x.f = f;
    unsigned u = x.u;
    return (unsigned short)((u + 0x7FFFu + ((u >> 16) & 1u)) >> 16);
}

// ---- TDM: tensor_load_to_lds of a 2D tile, with LDS row padding ----
__device__ __forceinline__ void tdm_load(unsigned lds_addr, const void* gptr,
                                         unsigned dim0, unsigned dim1,
                                         unsigned tile0, unsigned tile1,
                                         unsigned stride0, unsigned ds_code,
                                         unsigned pad_itv, unsigned pad_amt) {
    unsigned long long ga = (unsigned long long)gptr;
    v4u g0;
    g0[0] = 1u;                                   // count=1 valid user descriptor
    g0[1] = lds_addr;                             // LDS byte address
    g0[2] = (unsigned)ga;                         // global_addr[31:0]
    g0[3] = (unsigned)((ga >> 32) & 0x1FFFFFFu) | (2u << 30);  // addr[56:32] | type=2
    v8i g1;
    g1[0] = (int)((ds_code << 16) | (1u << 20) | (pad_itv << 22) | (pad_amt << 25));
    g1[1] = (int)((dim0 & 0xFFFFu) << 16);                         // tensor_dim0 lo
    g1[2] = (int)((dim0 >> 16) | ((dim1 & 0xFFFFu) << 16));        // dim0 hi | dim1 lo
    g1[3] = (int)((dim1 >> 16) | (tile0 << 16));                   // dim1 hi | tile_dim0
    g1[4] = (int)tile1;                                            // tile_dim1, tile_dim2=0
    g1[5] = (int)stride0;                                          // tensor_dim0_stride lo
    g1[6] = 0;
    g1[7] = 0;
    v4i g2 = {0, 0, 0, 0};
    v4i g3 = {0, 0, 0, 0};
    v8i g4 = {0, 0, 0, 0, 0, 0, 0, 0};
    __builtin_amdgcn_tensor_load_to_lds(g0, g1, g2, g3, g4, 0);
}

// ---------- degree / dinv ----------
__global__ void k_deg_init(float* deg, int n) {
    int i = blockIdx.x * blockDim.x + threadIdx.x;
    if (i < n) deg[i] = 1.0f;  // self loop
}

__global__ void k_deg_edges(const int* __restrict__ ei, float* deg, int E) {
    int stride = gridDim.x * blockDim.x;
    for (int e = blockIdx.x * blockDim.x + threadIdx.x; e < E; e += stride) {
        atomicAdd(&deg[ei[E + e]], 1.0f);
    }
}

__global__ void k_dinv(float* deg, int n) {
    int i = blockIdx.x * blockDim.x + threadIdx.x;
    if (i < n) deg[i] = rsqrtf(deg[i]);
}

// ---------- W1^T -> bf16 padded [NPAD][KPAD] ----------
__global__ void k_prep_w1(const float* __restrict__ W1, unsigned short* __restrict__ Wt) {
    int stride = gridDim.x * blockDim.x;
    for (int i = blockIdx.x * blockDim.x + threadIdx.x; i < NPAD * KPAD; i += stride) {
        int c = i / KPAD, k = i - c * KPAD;
        Wt[i] = (c < HID && k < F_IN) ? f2bf_bits(W1[(long)k * HID + c]) : (unsigned short)0;
    }
}

// ---------- GEMM1: h1[N][80] = x @ W1  (TDM double-buffer + bf16 WMMA) ----------
__global__ __launch_bounds__(256) void k_gemm1(const float* __restrict__ x,
                                               const unsigned short* __restrict__ Wt,
                                               float* __restrict__ h1, int n) {
    __shared__ __align__(16) float          Abuf[2][128 * ASTRIDE];  // 36 KB
    __shared__ __align__(16) unsigned short Bbuf[2][NPAD * BSTRIDE]; // 12.5 KB
    const int tid  = threadIdx.x;
    const int lane = tid & 31;
    const int wave = tid >> 5;
    const int lrow = lane & 15;
    const int hi   = lane >> 4;
    const int row0 = blockIdx.x * 128;
    const bool storeOK = (row0 + wave * 16 + 15) < n;

    v8f acc[5];
    #pragma unroll
    for (int t = 0; t < 5; ++t) acc[t] = {};

    if (wave == 0) {   // prologue: chunk 0 -> buffer 0 (EXEC all-ones in wave 0)
        tdm_load((unsigned)(uintptr_t)&Abuf[0][0], x + (size_t)row0 * F_IN,
                 F_IN, (unsigned)(n - row0), 32, 128, F_IN, /*f32*/2, /*32dw*/4, /*4dw*/3);
        tdm_load((unsigned)(uintptr_t)&Bbuf[0][0], Wt,
                 KPAD, NPAD, 32, NPAD, KPAD, /*bf16*/1, /*16dw*/3, /*4dw*/3);
    }

    for (int kc = 0; kc < KPAD; kc += 32) {
        const int pb = (kc >> 5) & 1;
        if (wave == 0) __builtin_amdgcn_s_wait_tensorcnt(0);
        __syncthreads();                       // chunk kc resident; all done with pb^1
        if (wave == 0 && kc + 32 < KPAD) {     // prefetch next chunk into other buffer
            const int kn = kc + 32;
            tdm_load((unsigned)(uintptr_t)&Abuf[pb ^ 1][0],
                     x + (size_t)row0 * F_IN + kn,
                     (unsigned)(F_IN - kn), (unsigned)(n - row0),
                     32, 128, F_IN, 2, 4, 3);
            tdm_load((unsigned)(uintptr_t)&Bbuf[pb ^ 1][0], Wt + kn,
                     (unsigned)(KPAD - kn), NPAD, 32, NPAD, KPAD, 1, 3, 3);
        }

        // A fragment: lane<16 -> K {0..7,16..23}; lane>=16 -> K {8..15,24..31}
        const float* Ar = &Abuf[pb][(wave * 16 + lrow) * ASTRIDE + (hi ? 8 : 0)];
        BF16x16 ua;
        #pragma unroll
        for (int i = 0; i < 4; ++i) {
            ua.h[i]      = tobf(Ar[i]);
            ua.h[4 + i]  = tobf(Ar[4 + i]);
            ua.h[8 + i]  = tobf(Ar[16 + i]);
            ua.h[12 + i] = tobf(Ar[20 + i]);
        }

        #pragma unroll
        for (int t = 0; t < 5; ++t) {
            // B fragment: col = 16t + lrow; K {8h..8h+7, 16+8h..16+8h+7} -> 2x ds_load_b128
            const int off = (t * 16 + lrow) * BSTRIDE + hi * 8;
            BF16x16 ub;
            ub.q[0] = *(const v8us*)&Bbuf[pb][off];
            ub.q[1] = *(const v8us*)&Bbuf[pb][off + 16];
            acc[t] = __builtin_amdgcn_wmma_f32_16x16x32_bf16(
                false, ua.v, false, ub.v, (short)0, acc[t], false, false);
        }
    }

    if (storeOK) {
        #pragma unroll
        for (int t = 0; t < 5; ++t) {
            #pragma unroll
            for (int i = 0; i < 8; ++i) {
                int r = row0 + wave * 16 + i + 8 * hi;   // C/D: M = vgpr + 8*(lane>>4)
                h1[(long)r * NPAD + t * 16 + lrow] = acc[t][i];
            }
        }
    }
}

// ---------- aggregation: self-loop init + edge scatter ----------
__global__ void k_agg_init(const float* __restrict__ dinv, const float* __restrict__ h,
                           float* __restrict__ out, int n, int width) {
    long total = (long)n * width;
    long stride = (long)gridDim.x * blockDim.x;
    for (long i = blockIdx.x * (long)blockDim.x + threadIdx.x; i < total; i += stride) {
        int r = (int)(i / width);
        float d = dinv[r];
        out[i] = d * d * h[i];
    }
}

__global__ void k_agg_edges1(const int* __restrict__ ei, const float* __restrict__ dinv,
                             const float* __restrict__ h1, float* out1, int E) {
    int warp  = (blockIdx.x * blockDim.x + threadIdx.x) >> 5;
    int lane  = threadIdx.x & 31;
    int nwarp = (gridDim.x * blockDim.x) >> 5;
    for (int e = warp; e < E; e += nwarp) {
        int src = ei[e], dst = ei[E + e];
        float norm = dinv[src] * dinv[dst];
        for (int c = lane; c < HID; c += 32)
            atomicAdd(&out1[(long)dst * NPAD + c], h1[(long)src * NPAD + c] * norm);
    }
}

__global__ void k_agg_edges2(const int* __restrict__ ei, const float* __restrict__ dinv,
                             const float* __restrict__ h2, float* out2, int E) {
    long total = (long)E * 8;
    long stride = (long)gridDim.x * blockDim.x;
    for (long i = blockIdx.x * (long)blockDim.x + threadIdx.x; i < total; i += stride) {
        int e = (int)(i >> 3), c = (int)(i & 7);
        if (c < NCLS) {
            int src = ei[e], dst = ei[E + e];
            float norm = dinv[src] * dinv[dst];
            atomicAdd(&out2[(long)dst * 8 + c], h2[(long)src * 8 + c] * norm);
        }
    }
}

// ---------- GEMM2: h2[N][8] = leaky_relu(out1 + b1) @ W2  (bf16 WMMA) ----------
__global__ __launch_bounds__(256) void k_gemm2(const float* __restrict__ out1,
                                               const float* __restrict__ b1,
                                               const float* __restrict__ W2,
                                               float* __restrict__ h2, int n) {
    const int tid  = threadIdx.x;
    const int lane = tid & 31;
    const int wave = tid >> 5;
    const int lrow = lane & 15;
    const int hi   = lane >> 4;
    const int rowbase = blockIdx.x * 128 + wave * 16;
    if (rowbase >= n) return;                      // wave-uniform exit, no barriers here
    int row = rowbase + lrow;
    if (row >= n) row = n - 1;
    const float* orow = out1 + (long)row * NPAD;

    v8f acc = {};
    #pragma unroll
    for (int kc = 0; kc < 96; kc += 32) {
        BF16x16 ua, ub;
        const int o1 = kc + (hi ? 8 : 0);
        const int o2 = kc + (hi ? 24 : 16);
        #pragma unroll
        for (int i = 0; i < 8; ++i) {
            int k = o1 + i;
            float v = 0.f;
            if (k < HID) { v = orow[k] + b1[k]; v = v > 0.f ? v : 0.01f * v; }
            ua.h[i] = tobf(v);
        }
        #pragma unroll
        for (int i = 0; i < 8; ++i) {
            int k = o2 + i;
            float v = 0.f;
            if (k < HID) { v = orow[k] + b1[k]; v = v > 0.f ? v : 0.01f * v; }
            ua.h[8 + i] = tobf(v);
        }
        const int kb = kc + hi * 8;
        #pragma unroll
        for (int e = 0; e < 16; ++e) {
            int k = kb + (e < 8 ? e : e + 8);
            ub.h[e] = (lrow < NCLS && k < HID) ? tobf(W2[(long)k * NCLS + lrow])
                                               : tobf(0.f);
        }
        acc = __builtin_amdgcn_wmma_f32_16x16x32_bf16(
            false, ua.v, false, ub.v, (short)0, acc, false, false);
    }

    if (lrow < 8) {
        #pragma unroll
        for (int i = 0; i < 8; ++i) {
            int r = rowbase + i + 8 * hi;
            if (r < n) h2[(long)r * 8 + lrow] = acc[i];
        }
    }
}

// ---------- log_softmax over 7 classes ----------
__global__ void k_logsoftmax(const float* __restrict__ out2, const float* __restrict__ b2,
                             float* __restrict__ y, int n) {
    int i = blockIdx.x * blockDim.x + threadIdx.x;
    if (i >= n) return;
    float z[NCLS], m = -INFINITY;
    #pragma unroll
    for (int c = 0; c < NCLS; ++c) { z[c] = out2[(long)i * 8 + c] + b2[c]; m = fmaxf(m, z[c]); }
    float s = 0.f;
    #pragma unroll
    for (int c = 0; c < NCLS; ++c) s += __expf(z[c] - m);
    float lse = __logf(s) + m;
    #pragma unroll
    for (int c = 0; c < NCLS; ++c) y[(long)i * NCLS + c] = z[c] - lse;
}

extern "C" void kernel_launch(void* const* d_in, const int* in_sizes, int n_in,
                              void* d_out, int out_size, void* d_ws, size_t ws_size,
                              hipStream_t stream) {
    const float* x  = (const float*)d_in[0];
    const int*   ei = (const int*)d_in[1];
    const float* W1 = (const float*)d_in[2];
    const float* b1 = (const float*)d_in[3];
    const float* W2 = (const float*)d_in[4];
    const float* b2 = (const float*)d_in[5];
    float* y = (float*)d_out;

    const int n = in_sizes[0] / F_IN;
    const int E = in_sizes[1] / 2;

    auto al = [](size_t v) { return (v + 255) & ~(size_t)255; };
    char* ws = (char*)d_ws;
    size_t off = 0;
    float* dinv = (float*)(ws + off);                 off += al((size_t)n * 4);
    unsigned short* Wt = (unsigned short*)(ws + off); off += al((size_t)NPAD * KPAD * 2);
    float* h1   = (float*)(ws + off);                 off += al((size_t)n * NPAD * 4);
    float* out1 = (float*)(ws + off);                 off += al((size_t)n * NPAD * 4);
    float* h2   = (float*)(ws + off);                 off += al((size_t)n * 8 * 4);
    float* out2 = (float*)(ws + off);                 off += al((size_t)n * 8 * 4);

    const int T = 256;
    int gN   = (n + T - 1) / T;
    int gE   = min((E + T - 1) / T, 65535);
    int gEw  = min((E * 32 + T - 1) / T, 131072);       // warp per edge
    int gE8  = min((E * 8 + T - 1) / T, 131072);
    int strips = (n + 15) / 16;
    int gG   = (strips + 7) / 8;                         // 8 waves * 16 rows per block
    long tot1 = (long)n * NPAD;
    int gI1  = (int)min((tot1 + T - 1) / T, (long)65535);
    long tot2 = (long)n * 8;
    int gI2  = (int)min((tot2 + T - 1) / T, (long)65535);

    k_deg_init  <<<gN,  T, 0, stream>>>(dinv, n);
    k_deg_edges <<<gE,  T, 0, stream>>>(ei, dinv, E);
    k_dinv      <<<gN,  T, 0, stream>>>(dinv, n);
    k_prep_w1   <<<min((NPAD * KPAD + T - 1) / T, 4096), T, 0, stream>>>(W1, Wt);
    k_gemm1     <<<gG,  T, 0, stream>>>(x, Wt, h1, n);
    k_agg_init  <<<gI1, T, 0, stream>>>(dinv, h1, out1, n, NPAD);
    k_agg_edges1<<<gEw, T, 0, stream>>>(ei, dinv, h1, out1, E);
    k_gemm2     <<<gG,  T, 0, stream>>>(out1, b1, W2, h2, n);
    k_agg_init  <<<gI2, T, 0, stream>>>(dinv, h2, out2, n, 8);
    k_agg_edges2<<<gE8, T, 0, stream>>>(ei, dinv, h2, out2, E);
    k_logsoftmax<<<gN,  T, 0, stream>>>(out2, b2, y, n);
}
